// CrossAttention_9637906612679
// MI455X (gfx1250) — matrix operand order repaired and checked by reading, not compiled, not run on hardware
//
#include <hip/hip_runtime.h>
#include <stdint.h>

// ---------------------------------------------------------------------------
// Types for CDNA5 WMMA (wave32, 16x16x32 bf16 -> f32)
// ---------------------------------------------------------------------------
typedef __bf16 bf16_t;
typedef bf16_t v16bf __attribute__((ext_vector_type(16)));
typedef float  v8f   __attribute__((ext_vector_type(8)));

#define DEVINL __device__ __forceinline__

// round-to-nearest-even f32 -> bf16 (stored as u16)
DEVINL unsigned short f2bf(float f) {
    union { float f; uint32_t u; } v; v.f = f;
    uint32_t u = v.u;
    u += 0x7FFFu + ((u >> 16) & 1u);
    return (unsigned short)(u >> 16);
}

union Frag16 {
    v16bf v;
    uint4 q[2];
    unsigned short h[16];
};

// A-fragment: halves 0..7 from p0 (contiguous), halves 8..15 from p1 (contiguous)
DEVINL v16bf ld_frag_2x8(const unsigned short* p0, const unsigned short* p1) {
    Frag16 f;
    f.q[0] = *reinterpret_cast<const uint4*>(p0);
    f.q[1] = *reinterpret_cast<const uint4*>(p1);
    return f.v;
}
// B-fragment: 16 contiguous halves
DEVINL v16bf ld_frag_16(const unsigned short* p) {
    Frag16 f;
    f.q[0] = *reinterpret_cast<const uint4*>(p);
    f.q[1] = *reinterpret_cast<const uint4*>(p + 8);
    return f.v;
}

DEVINL v8f wmma_bf16(v16bf a, v16bf b, v8f c) {
    return __builtin_amdgcn_wmma_f32_16x16x32_bf16(
        /*neg_a=*/false, a, /*neg_b=*/false, b,
        /*c_mod=*/(short)0, c, /*reuse_a=*/false, /*reuse_b=*/false);
}

// ---------------------------------------------------------------------------
// Async global -> LDS copy (gfx1250, ASYNCcnt tracked, no VGPR data movement)
// lds destination given as wave-relative LDS byte address (= low 32 bits of
// the generic pointer to a __shared__ object, per the LDS aperture rule).
// ---------------------------------------------------------------------------
DEVINL void async_cp_b128(const void* gsrc, void* lds_dst) {
    uint32_t lds_addr = (uint32_t)(uintptr_t)lds_dst;
    uint64_t gaddr    = (uint64_t)(uintptr_t)gsrc;
    asm volatile("global_load_async_to_lds_b128 %0, %1, off"
                 :: "v"(lds_addr), "v"(gaddr)
                 : "memory");
}
DEVINL void wait_async0() {
    asm volatile("s_wait_asynccnt 0" ::: "memory");
}

// ---------------------------------------------------------------------------
// Problem constants
// ---------------------------------------------------------------------------
static constexpr int Bsz   = 4;
static constexpr int NQ    = 2048;
static constexpr int NKV   = 2048;
static constexpr int DMOD  = 1024;
static constexpr int NH    = 16;
static constexpr int DK    = 64;
static constexpr int MTOT  = Bsz * NQ;   // 8192 rows for the big GEMMs

// ---------------------------------------------------------------------------
// f32 -> bf16 conversion (vectorized, n must be multiple of 4)
// ---------------------------------------------------------------------------
__global__ __launch_bounds__(256) void cvt_f32_bf16(const float* __restrict__ in,
                                                    unsigned short* __restrict__ out,
                                                    int n4) {
    int i = blockIdx.x * blockDim.x + threadIdx.x;
    if (i < n4) {
        float4 f = reinterpret_cast<const float4*>(in)[i];
        union { unsigned short h[4]; uint2 u; } o;
        o.h[0] = f2bf(f.x); o.h[1] = f2bf(f.y);
        o.h[2] = f2bf(f.z); o.h[3] = f2bf(f.w);
        reinterpret_cast<uint2*>(out)[i] = o.u;
    }
}

// ---------------------------------------------------------------------------
// bf16 GEMM: C[8192,1024] = A[8192,1024] @ B[1024,1024]
// 256 threads = 8 waves, block tile 128x128, K-step 32.
// Wave (wr,wc): wr in 0..3 owns 32 rows, wc in 0..1 owns 64 cols.
// A tile staged with async global->LDS DMA; B tile staged transposed.
// EPI==0: write bf16 head-split [B][H][Ntok][64]; EPI==1: write f32 row-major.
// ---------------------------------------------------------------------------
template<int EPI>
__global__ __launch_bounds__(256) void gemm_bf16_128x128(
    const unsigned short* __restrict__ A,
    const unsigned short* __restrict__ Bm,
    void* __restrict__ Cout)
{
    constexpr int K = DMOD, N = DMOD;
    __shared__ __align__(16) unsigned short Atile[128 * 32];
    __shared__ __align__(16) unsigned short Btile[128 * 32];   // transposed: [col][k]

    const int t    = threadIdx.x;
    const int lane = t & 31;
    const int wave = t >> 5;
    const int wr   = wave & 3;
    const int wc   = wave >> 2;
    const int lg   = lane >> 4;    // lane group (0/1)
    const int ln   = lane & 15;
    const int bm   = blockIdx.x * 128;
    const int bn   = blockIdx.y * 128;

    v8f acc[2][4];
    #pragma unroll
    for (int i = 0; i < 2; ++i)
        #pragma unroll
        for (int j = 0; j < 4; ++j) acc[i][j] = {};

    for (int k0 = 0; k0 < K; k0 += 32) {
        // ---- stage A tile [128][32] : 512 x b128 async DMA, 2 per thread
        #pragma unroll
        for (int i = 0; i < 2; ++i) {
            int v   = t + i * 256;
            int row = v >> 2, cc = (v & 3) * 8;
            async_cp_b128(&A[(bm + row) * K + k0 + cc], &Atile[row * 32 + cc]);
        }
        // ---- stage B tile transposed [128 cols][32 k] (element-wise)
        #pragma unroll
        for (int i = 0; i < 16; ++i) {
            int idx = t + i * 256;
            int c = idx & 127, kk = idx >> 7;
            Btile[c * 32 + kk] = Bm[(k0 + kk) * N + bn + c];
        }
        if (k0 + 32 < K) {
            // prefetch next B tile rows (gfx1250 global_prefetch_b8)
            __builtin_prefetch(&Bm[(k0 + 32 + (t >> 3)) * N + bn + (t & 7) * 16], 0, 1);
        }
        wait_async0();
        __syncthreads();

        // ---- A fragments (2 m-subtiles)
        v16bf af[2];
        #pragma unroll
        for (int mi = 0; mi < 2; ++mi) {
            int row = wr * 32 + mi * 16 + ln;
            const unsigned short* base = &Atile[row * 32 + lg * 16];
            af[mi] = ld_frag_2x8(base, base + 8);
        }
        // ---- B fragments + WMMA
        #pragma unroll
        for (int ni = 0; ni < 4; ++ni) {
            int col = wc * 64 + ni * 16 + ln;
            v16bf bfr = ld_frag_16(&Btile[col * 32 + lg * 16]);
            #pragma unroll
            for (int mi = 0; mi < 2; ++mi)
                acc[mi][ni] = wmma_bf16(af[mi], bfr, acc[mi][ni]);
        }
        __syncthreads();
    }

    // ---- epilogue
    #pragma unroll
    for (int mi = 0; mi < 2; ++mi) {
        #pragma unroll
        for (int ni = 0; ni < 4; ++ni) {
            int n = bn + wc * 64 + ni * 16 + ln;
            #pragma unroll
            for (int r = 0; r < 8; ++r) {
                int m = bm + wr * 32 + mi * 16 + r + 8 * lg;
                float vv = acc[mi][ni][r];
                if (EPI == 0) {
                    int b = m >> 11, tok = m & (NQ - 1);
                    int h = n >> 6,  dk  = n & (DK - 1);
                    ((unsigned short*)Cout)[(((size_t)(b * NH + h) * NQ + tok) * DK) + dk] =
                        f2bf(vv);
                } else {
                    ((float*)Cout)[(size_t)m * DMOD + n] = vv;
                }
            }
        }
    }
}

// ---------------------------------------------------------------------------
// Flash-style attention per (b,h): ctx = softmax(Q K^T + bias) V
// Q,K,V: bf16 [B][H][N][64].  bias: f32 [H][NQ][NKV] (broadcast over B).
// ctx out: bf16 [B][NQ][1024] (head-merged), ready for the Wo GEMM.
// 128 threads = 4 waves; block = 64 q rows; each wave owns 16 q rows.
// K tile staged with async global->LDS DMA; V tile staged transposed.
// ---------------------------------------------------------------------------
__global__ __launch_bounds__(128) void flash_attn_bf16(
    const unsigned short* __restrict__ Qh,
    const unsigned short* __restrict__ Kh,
    const unsigned short* __restrict__ Vh,
    const float* __restrict__ bias,
    unsigned short* __restrict__ ctx)
{
    __shared__ __align__(16) unsigned short Kt[64 * 64];        // [kv][dk]
    __shared__ __align__(16) unsigned short Vt[64 * 64];        // [dk][kv] (transposed)
    __shared__ __align__(16) unsigned short Pl[4][16 * 64];     // per-wave P tile

    const int t    = threadIdx.x;
    const int lane = t & 31;
    const int wave = t >> 5;
    const int lg   = lane >> 4;
    const int ln   = lane & 15;
    const int bh   = blockIdx.y;          // b*NH + h
    const int b    = bh >> 4;
    const int h    = bh & 15;
    const int q0   = blockIdx.x * 64;
    const int qw   = q0 + wave * 16;      // this wave's q rows

    const unsigned short* Qp = Qh + (size_t)bh * NQ * DK;
    const unsigned short* Kp = Kh + (size_t)bh * NKV * DK;
    const unsigned short* Vp = Vh + (size_t)bh * NKV * DK;
    const float*          bp = bias + ((size_t)h * NQ + qw) * NKV;

    // Q fragments: 2 K-chunks of 32, reused across the whole kv loop
    v16bf qf[2];
    {
        const unsigned short* qr = Qp + (qw + ln) * DK;
        #pragma unroll
        for (int tc = 0; tc < 2; ++tc) {
            const unsigned short* p0 = qr + tc * 32 + lg * 16;
            qf[tc] = ld_frag_2x8(p0, p0 + 8);
        }
    }

    v8f cacc[4];
    #pragma unroll
    for (int d = 0; d < 4; ++d) cacc[d] = {};
    float mrow[8], lrow[8];
    #pragma unroll
    for (int r = 0; r < 8; ++r) { mrow[r] = -1.0e30f; lrow[r] = 0.0f; }

    for (int kv0 = 0; kv0 < NKV; kv0 += 64) {
        // ---- stage K tile row-major: 512 x b128 async DMA, 4 per thread
        #pragma unroll
        for (int i = 0; i < 4; ++i) {
            int v = t + i * 128;
            int row = v >> 3, cc = (v & 7) * 8;
            async_cp_b128(&Kp[(kv0 + row) * DK + cc], &Kt[row * 64 + cc]);
        }
        // ---- stage V tile transposed [dk][kv] (element-wise)
        #pragma unroll
        for (int i = 0; i < 32; ++i) {
            int idx = t + i * 128;
            int dk = idx & 63, kv = idx >> 6;
            Vt[dk * 64 + kv] = Vp[(kv0 + kv) * DK + dk];
        }
        wait_async0();
        __syncthreads();

        // ---- S = position_bias (loaded straight into the accumulator)
        v8f s[4];
        #pragma unroll
        for (int j = 0; j < 4; ++j) {
            #pragma unroll
            for (int r = 0; r < 8; ++r) {
                int qrow = r + 8 * lg;
                s[j][r] = bp[(size_t)qrow * NKV + kv0 + j * 16 + ln];
            }
        }
        // ---- S += Q K^T via WMMA (B column n = K row kv0+16j+n, contiguous)
        #pragma unroll
        for (int j = 0; j < 4; ++j) {
            #pragma unroll
            for (int tc = 0; tc < 2; ++tc) {
                v16bf bfr = ld_frag_16(&Kt[(j * 16 + ln) * 64 + tc * 32 + lg * 16]);
                s[j] = wmma_bf16(qf[tc], bfr, s[j]);
            }
        }

        // ---- online softmax over this 64-wide tile
        #pragma unroll
        for (int r = 0; r < 8; ++r) {
            float mx = fmaxf(fmaxf(s[0][r], s[1][r]), fmaxf(s[2][r], s[3][r]));
            #pragma unroll
            for (int m2 = 1; m2 < 16; m2 <<= 1)
                mx = fmaxf(mx, __shfl_xor(mx, m2, 32));
            float newm  = fmaxf(mrow[r], mx);
            float scale = __expf(mrow[r] - newm);
            mrow[r] = newm;
            float sum = 0.0f;
            #pragma unroll
            for (int j = 0; j < 4; ++j) {
                float p = __expf(s[j][r] - newm);
                s[j][r] = p;
                sum += p;
            }
            #pragma unroll
            for (int m2 = 1; m2 < 16; m2 <<= 1)
                sum += __shfl_xor(sum, m2, 32);
            lrow[r] = lrow[r] * scale + sum;
            #pragma unroll
            for (int d = 0; d < 4; ++d) cacc[d][r] *= scale;
        }

        // ---- write P (bf16) to per-wave LDS, re-read as A fragments
        unsigned short* pw = &Pl[wave][0];
        #pragma unroll
        for (int j = 0; j < 4; ++j)
            #pragma unroll
            for (int r = 0; r < 8; ++r)
                pw[(r + 8 * lg) * 64 + j * 16 + ln] = f2bf(s[j][r]);

        // ---- ctx += P @ V (same-wave LDS ops are in-order)
        #pragma unroll
        for (int tc = 0; tc < 2; ++tc) {
            const unsigned short* pr = &Pl[wave][ln * 64 + tc * 32 + lg * 16];
            v16bf af = ld_frag_2x8(pr, pr + 8);
            #pragma unroll
            for (int d = 0; d < 4; ++d) {
                v16bf bfr = ld_frag_16(&Vt[(d * 16 + ln) * 64 + tc * 32 + lg * 16]);
                cacc[d] = wmma_bf16(af, bfr, cacc[d]);
            }
        }
        __syncthreads();
    }

    // ---- normalize and store head-merged bf16 ctx [B][NQ][1024]
    #pragma unroll
    for (int d = 0; d < 4; ++d) {
        #pragma unroll
        for (int r = 0; r < 8; ++r) {
            int qrow = qw + r + 8 * lg;
            float vv = cacc[d][r] / lrow[r];
            ctx[((size_t)(b * NQ + qrow)) * DMOD + h * DK + d * 16 + ln] = f2bf(vv);
        }
    }
}

// ---------------------------------------------------------------------------
// Host-side orchestration
// ---------------------------------------------------------------------------
extern "C" void kernel_launch(void* const* d_in, const int* in_sizes, int n_in,
                              void* d_out, int out_size, void* d_ws, size_t ws_size,
                              hipStream_t stream) {
    const float* x    = (const float*)d_in[0];
    const float* enc  = (const float*)d_in[1];
    const float* pbias= (const float*)d_in[2];
    const float* Wq   = (const float*)d_in[3];
    const float* Wk   = (const float*)d_in[4];
    const float* Wv   = (const float*)d_in[5];
    const float* Wo   = (const float*)d_in[6];
    float*       out  = (float*)d_out;

    const size_t actE = (size_t)MTOT * DMOD;       // 8,388,608
    const size_t wE   = (size_t)DMOD * DMOD;       // 1,048,576

    char* ws = (char*)d_ws;
    size_t off = 0;
    auto alloc = [&](size_t bytes) -> void* {
        void* p = ws + off;
        off += (bytes + 255) & ~(size_t)255;
        return p;
    };
    unsigned short* xb   = (unsigned short*)alloc(actE * 2);
    unsigned short* eb   = (unsigned short*)alloc(actE * 2);
    unsigned short* wqb  = (unsigned short*)alloc(wE * 2);
    unsigned short* wkb  = (unsigned short*)alloc(wE * 2);
    unsigned short* wvb  = (unsigned short*)alloc(wE * 2);
    unsigned short* wob  = (unsigned short*)alloc(wE * 2);
    unsigned short* Qb   = (unsigned short*)alloc(actE * 2);  // [B][H][NQ][64]
    unsigned short* Kb   = (unsigned short*)alloc(actE * 2);
    unsigned short* Vb   = (unsigned short*)alloc(actE * 2);
    unsigned short* ctxb = (unsigned short*)alloc(actE * 2);  // [B][NQ][1024]

    // --- f32 -> bf16 conversions
    {
        int n4a = (int)(actE / 4), n4w = (int)(wE / 4);
        cvt_f32_bf16<<<(n4a + 255) / 256, 256, 0, stream>>>(x,   xb,  n4a);
        cvt_f32_bf16<<<(n4a + 255) / 256, 256, 0, stream>>>(enc, eb,  n4a);
        cvt_f32_bf16<<<(n4w + 255) / 256, 256, 0, stream>>>(Wq,  wqb, n4w);
        cvt_f32_bf16<<<(n4w + 255) / 256, 256, 0, stream>>>(Wk,  wkb, n4w);
        cvt_f32_bf16<<<(n4w + 255) / 256, 256, 0, stream>>>(Wv,  wvb, n4w);
        cvt_f32_bf16<<<(n4w + 255) / 256, 256, 0, stream>>>(Wo,  wob, n4w);
    }

    // --- Q/K/V projections (head-split bf16 epilogue)
    dim3 ggrid(MTOT / 128, DMOD / 128);
    gemm_bf16_128x128<0><<<ggrid, 256, 0, stream>>>(xb, wqb, Qb);
    gemm_bf16_128x128<0><<<ggrid, 256, 0, stream>>>(eb, wkb, Kb);
    gemm_bf16_128x128<0><<<ggrid, 256, 0, stream>>>(eb, wvb, Vb);

    // --- flash attention with additive position bias
    dim3 agrid(NQ / 64, Bsz * NH);
    flash_attn_bf16<<<agrid, 128, 0, stream>>>(Qb, Kb, Vb, pbias, ctxb);

    // --- output projection, f32 epilogue to d_out
    gemm_bf16_128x128<1><<<ggrid, 256, 0, stream>>>(ctxb, wob, out);
}
